// GeometricGAT_54812372631799
// MI455X (gfx1250) — matrix operand order repaired and checked by reading, not compiled, not run on hardware
//
#include <hip/hip_runtime.h>
#include <hip/hip_bf16.h>
#include <stdint.h>

// ---------------------------------------------------------------------------
// GATv2 x2 on gfx1250: LDS-staged bf16 WMMA GEMMs (1x4 register blocking,
// back-to-back WMMA issue) + vectorized atomic segment softmax/aggregation.
// ---------------------------------------------------------------------------

#define NEG_SLOPE 0.2f

typedef __attribute__((ext_vector_type(16))) __bf16 v16bf;
typedef __attribute__((ext_vector_type(8)))  float  v8f;
typedef __attribute__((ext_vector_type(4)))  float  f32x4;

union Frag16 { v16bf v; f32x4 x[2]; };

__device__ __forceinline__ void atomAddF(float* p, float v) {
  unsafeAtomicAdd(p, v);   // lowers to global_atomic_add_f32
}

// order-preserving float <-> uint encoding for atomicMax-based segment max
__device__ __forceinline__ unsigned encF(float f) {
  unsigned u = __float_as_uint(f);
  return (u & 0x80000000u) ? ~u : (u | 0x80000000u);
}
__device__ __forceinline__ float decF(unsigned u) {
  u = (u & 0x80000000u) ? (u & 0x7FFFFFFFu) : ~u;
  return __uint_as_float(u);
}

// ---------------------------------------------------------------------------
// self-loop attr = mean of incoming edge attrs
// ---------------------------------------------------------------------------
__global__ void gat_degree_kernel(const int* __restrict__ ei,
                                  const float* __restrict__ eattr,
                                  float* __restrict__ cnt, float* __restrict__ sum,
                                  int E) {
  int e = blockIdx.x * blockDim.x + threadIdx.x;
  if (e < E) {
    int d = ei[E + e];
    atomAddF(&cnt[d], 1.0f);
    atomAddF(&sum[d], eattr[e]);
  }
}

__global__ void gat_loopattr_kernel(const float* __restrict__ cnt,
                                    const float* __restrict__ sum,
                                    float* __restrict__ la, int N) {
  int n = blockIdx.x * blockDim.x + threadIdx.x;
  if (n < N) la[n] = sum[n] / fmaxf(cnt[n], 1.0f);
}

// ---------------------------------------------------------------------------
// fp32 -> bf16 convert; weight convert+transpose ([K,M] f32 -> [M,K] bf16)
// ---------------------------------------------------------------------------
__global__ void gat_cvt_bf16_kernel(const float* __restrict__ in,
                                    __bf16* __restrict__ out, int n) {
  int i = blockIdx.x * blockDim.x + threadIdx.x;
  if (i < n) out[i] = (__bf16)in[i];
}

__global__ void gat_transpose_cvt_kernel(const float* __restrict__ W,
                                         __bf16* __restrict__ Wt, int K, int M) {
  int i = blockIdx.x * blockDim.x + threadIdx.x;
  if (i < K * M) {
    int k = i / M;
    int m = i - k * M;
    Wt[(size_t)m * K + k] = (__bf16)W[i];
  }
}

// ---------------------------------------------------------------------------
// C[N,M] = A[N,K](bf16) @ Bt[M,K]^T(bf16) + bias.
// Block = 256 threads (8 waves) -> 128 rows x 64 cols of C.
// W^T strip (64 cols x K) staged in LDS once per block (row pad 8 bf16 so
// the 16-lane B-fragment reads hit disjoint bank quads). Each wave holds one
// A fragment and 4 distinct B fragments so the 4 WMMAs issue back-to-back.
// Grid exact, no divergence -> EXEC all ones for WMMA. K must be <= 256.
// ---------------------------------------------------------------------------
#define GROWS 128
#define GCOLS 64
#define KMAX  256
#define BSTRIDE (KMAX + 8)   // bf16 elems; 528B row stride -> 4-bank lane step

__global__ void __launch_bounds__(256)
gat_gemm_wmma_bf16(const __bf16* __restrict__ A,
                   const __bf16* __restrict__ Bt,
                   const float* __restrict__ bias,
                   float* __restrict__ Cout,
                   int K, int M) {
  __shared__ __bf16 Bs[GCOLS * BSTRIDE];

  const int tid  = threadIdx.x;
  const int wave = tid >> 5;
  const int lane = tid & 31;
  const int half = lane >> 4;
  const int l16  = lane & 15;

  const int stripsM  = M / GCOLS;
  const int brow     = blockIdx.x / stripsM;
  const int bcolblk  = blockIdx.x - brow * stripsM;
  const int colbase  = bcolblk * GCOLS;

  // cooperative fill: 64 cols x K bf16 of W^T into LDS, 16B chunks
  {
    const int kchunks = K / 8;                    // f32x4 chunks per column
    const int total   = GCOLS * kchunks;
    for (int i = tid; i < total; i += 256) {
      int col = i / kchunks;
      int kc  = i - col * kchunks;
      *((f32x4*)(Bs + (size_t)col * BSTRIDE) + kc) =
          *((const f32x4*)(Bt + (size_t)(colbase + col) * K) + kc);
    }
  }
  __syncthreads();

  const int row = brow * GROWS + wave * 16 + l16;
  const __bf16* arow = A + (size_t)row * K;

  v8f acc0 = {}, acc1 = {}, acc2 = {}, acc3 = {};
  for (int kb = 0; kb < K; kb += 32) {
    Frag16 fa;
    // A 16x32 bf16 layout: elems 0..7  -> K = kb + half*8 + 0..7
    //                      elems 8..15 -> K = kb + 16 + half*8 + 0..7
    fa.x[0] = *(const f32x4*)(arow + kb + half * 8);
    fa.x[1] = *(const f32x4*)(arow + kb + 16 + half * 8);

    // B 32x16 bf16 layout per tile: elems 0..7 -> K = kb + half*8 + 0..7, elems 8..15 -> K = kb + 16 + half*8 + 0..7
    const __bf16* bbase = Bs + (size_t)l16 * BSTRIDE + kb + half * 8;
    Frag16 fb0, fb1, fb2, fb3;
    fb0.x[0] = *(const f32x4*)(bbase + (size_t)0 * 16 * BSTRIDE);
    fb0.x[1] = *(const f32x4*)(bbase + (size_t)0 * 16 * BSTRIDE + 16);
    fb1.x[0] = *(const f32x4*)(bbase + (size_t)1 * 16 * BSTRIDE);
    fb1.x[1] = *(const f32x4*)(bbase + (size_t)1 * 16 * BSTRIDE + 16);
    fb2.x[0] = *(const f32x4*)(bbase + (size_t)2 * 16 * BSTRIDE);
    fb2.x[1] = *(const f32x4*)(bbase + (size_t)2 * 16 * BSTRIDE + 16);
    fb3.x[0] = *(const f32x4*)(bbase + (size_t)3 * 16 * BSTRIDE);
    fb3.x[1] = *(const f32x4*)(bbase + (size_t)3 * 16 * BSTRIDE + 16);

    acc0 = __builtin_amdgcn_wmma_f32_16x16x32_bf16(
        false, fa.v, false, fb0.v, (short)0, acc0, false, false);
    acc1 = __builtin_amdgcn_wmma_f32_16x16x32_bf16(
        false, fa.v, false, fb1.v, (short)0, acc1, false, false);
    acc2 = __builtin_amdgcn_wmma_f32_16x16x32_bf16(
        false, fa.v, false, fb2.v, (short)0, acc2, false, false);
    acc3 = __builtin_amdgcn_wmma_f32_16x16x32_bf16(
        false, fa.v, false, fb3.v, (short)0, acc3, false, false);
  }

  float* crow = Cout + (size_t)(brow * GROWS + wave * 16 + 8 * half) * M;
#define GAT_STORE(T, ACC)                                                     \
  {                                                                           \
    const int col = colbase + (T) * 16 + l16;                                 \
    const float bv = bias[col];                                               \
    _Pragma("unroll")                                                         \
    for (int r = 0; r < 8; ++r) crow[(size_t)r * M + col] = ACC[r] + bv;      \
  }
  GAT_STORE(0, acc0)
  GAT_STORE(1, acc1)
  GAT_STORE(2, acc2)
  GAT_STORE(3, acc3)
#undef GAT_STORE
}

// ---------------------------------------------------------------------------
// logits[e,h] = sum_c att[h,c]*leakyrelu(xl[src,hc] + xr[dst,hc] + ea*We[hc])
// one wave per edge, f32x4 channel walk; atomicMax segment max.
// ---------------------------------------------------------------------------
__global__ void gat_logits_kernel(const float* __restrict__ xl,
                                  const float* __restrict__ xr,
                                  const int* __restrict__ ei,
                                  const float* __restrict__ eattr,
                                  const float* __restrict__ loopattr,
                                  const float* __restrict__ We,
                                  const float* __restrict__ att,
                                  float* __restrict__ logits,
                                  unsigned* __restrict__ mEnc,
                                  int E, int N, int H, int C) {
  const int wid  = (blockIdx.x * blockDim.x + threadIdx.x) >> 5;
  const int lane = threadIdx.x & 31;
  const int Etot = E + N;
  if (wid >= Etot) return;

  int srcN, dstN; float eav;
  if (wid < E) { srcN = ei[wid]; dstN = ei[E + wid]; eav = eattr[wid]; }
  else         { srcN = dstN = wid - E;              eav = loopattr[srcN]; }

  const int HC = H * C;
  const int Q  = HC >> 2;
  const f32x4* pl4 = (const f32x4*)(xl + (size_t)srcN * HC);
  const f32x4* pr4 = (const f32x4*)(xr + (size_t)dstN * HC);
  const f32x4* We4 = (const f32x4*)We;
  const f32x4* at4 = (const f32x4*)att;

  float hacc[4] = {0.f, 0.f, 0.f, 0.f};   // H <= 4
  for (int q = lane; q < Q; q += 32) {
    f32x4 a = pl4[q], b = pr4[q], w = We4[q], t = at4[q];
    float part = 0.f;
#pragma unroll
    for (int j = 0; j < 4; ++j) {
      float s = a[j] + b[j] + eav * w[j];
      s = (s > 0.f) ? s : NEG_SLOPE * s;
      part += s * t[j];
    }
    hacc[(q * 4) / C] += part;
  }
#pragma unroll
  for (int h = 0; h < 4; ++h) {
    if (h >= H) break;
    float v = hacc[h];
#pragma unroll
    for (int off = 16; off > 0; off >>= 1) v += __shfl_xor(v, off, 32);
    if (lane == 0) {
      logits[(size_t)wid * H + h] = v;
      atomicMax(&mEnc[(size_t)dstN * H + h], encF(v));
    }
  }
}

// ex = exp(logit - m[dst]); denom[dst] += ex
__global__ void gat_exp_kernel(const float* __restrict__ logits,
                               const int* __restrict__ ei,
                               const unsigned* __restrict__ mEnc,
                               float* __restrict__ ex,
                               float* __restrict__ denom,
                               int E, int N, int H) {
  int i = blockIdx.x * blockDim.x + threadIdx.x;
  int tot = (E + N) * H;
  if (i >= tot) return;
  int e = i / H, h = i - e * H;
  int dstN = (e < E) ? ei[E + e] : (e - E);
  float m = decF(mEnc[(size_t)dstN * H + h]);
  float v = __expf(logits[i] - m);
  ex[i] = v;
  atomAddF(&denom[(size_t)dstN * H + h], v);
}

// alpha = ex / (denom[dst] + 1e-16), in place
__global__ void gat_alpha_kernel(float* __restrict__ ex,
                                 const int* __restrict__ ei,
                                 const float* __restrict__ denom,
                                 int E, int N, int H) {
  int i = blockIdx.x * blockDim.x + threadIdx.x;
  int tot = (E + N) * H;
  if (i >= tot) return;
  int e = i / H, h = i - e * H;
  int dstN = (e < E) ? ei[E + e] : (e - E);
  ex[i] = ex[i] / (denom[(size_t)dstN * H + h] + 1e-16f);
}

// out[dst] += alpha * xl[src]; concat: out[dst, h*C+c]; mean: out[dst,c] += /H
__global__ void gat_aggregate_kernel(const float* __restrict__ xl,
                                     const float* __restrict__ alpha,
                                     const int* __restrict__ ei,
                                     float* __restrict__ out,
                                     int E, int N, int H, int C, int concat) {
  const int wid  = (blockIdx.x * blockDim.x + threadIdx.x) >> 5;
  const int lane = threadIdx.x & 31;
  const int Etot = E + N;
  if (wid >= Etot) return;

  int srcN, dstN;
  if (wid < E) { srcN = ei[wid]; dstN = ei[E + wid]; }
  else         { srcN = dstN = wid - E; }

  const int HC = H * C;
  const int Q  = HC >> 2;
  const f32x4* pl4 = (const f32x4*)(xl + (size_t)srcN * HC);
  const float scale = concat ? 1.0f : (1.0f / (float)H);
  float al[4];
#pragma unroll
  for (int h = 0; h < 4; ++h)
    al[h] = (h < H) ? scale * alpha[(size_t)wid * H + h] : 0.f;

  if (concat) {
    float* po = out + (size_t)dstN * HC;
    for (int q = lane; q < Q; q += 32) {
      f32x4 v = pl4[q];
      const int c0 = q * 4;
      const float a = al[c0 / C];
#pragma unroll
      for (int j = 0; j < 4; ++j) atomAddF(&po[c0 + j], a * v[j]);
    }
  } else {
    float* po = out + (size_t)dstN * C;
    for (int q = lane; q < Q; q += 32) {
      f32x4 v = pl4[q];
      const int c0 = q * 4;
      const float a = al[c0 / C];
      const int cb = c0 % C;
#pragma unroll
      for (int j = 0; j < 4; ++j) atomAddF(&po[cb + j], a * v[j]);
    }
  }
}

__global__ void gat_bias_relu_kernel(float* __restrict__ h,
                                     const float* __restrict__ b,
                                     int n, int M) {
  int i = blockIdx.x * blockDim.x + threadIdx.x;
  if (i < n) h[i] = fmaxf(h[i] + b[i % M], 0.0f);
}

__global__ void gat_init_out_kernel(float* __restrict__ out,
                                    const float* __restrict__ b,
                                    int n, int M) {
  int i = blockIdx.x * blockDim.x + threadIdx.x;
  if (i < n) out[i] = b[i % M];
}

// ---------------------------------------------------------------------------
// host side
// ---------------------------------------------------------------------------
static inline size_t rup(size_t x) { return (x + 255) & ~(size_t)255; }
static inline int cdiv(long long a, int b) { return (int)((a + b - 1) / b); }

extern "C" void kernel_launch(void* const* d_in, const int* in_sizes, int n_in,
                              void* d_out, int out_size, void* d_ws, size_t ws_size,
                              hipStream_t stream) {
  const float* x    = (const float*)d_in[0];
  const int*   ei   = (const int*)  d_in[1];
  const float* ea   = (const float*)d_in[2];
  const float* Wl1  = (const float*)d_in[3];
  const float* bl1  = (const float*)d_in[4];
  const float* Wr1  = (const float*)d_in[5];
  const float* br1  = (const float*)d_in[6];
  const float* We1  = (const float*)d_in[7];
  const float* att1 = (const float*)d_in[8];
  const float* b1   = (const float*)d_in[9];
  const float* Wl2  = (const float*)d_in[10];
  const float* bl2  = (const float*)d_in[11];
  const float* Wr2  = (const float*)d_in[12];
  const float* br2  = (const float*)d_in[13];
  const float* We2  = (const float*)d_in[14];
  const float* att2 = (const float*)d_in[15];
  const float* b2   = (const float*)d_in[16];

  const int D = 256, H = 4, C1 = 64, C2 = 256;
  const int M1 = H * C1;            // 256
  const int M2 = H * C2;            // 1024
  const int N = in_sizes[0] / D;    // 16384
  const int E = in_sizes[1] / 2;    // 131072
  const int Etot = E + N;           // 147456

  // ---- workspace carve-out ----
  char* ws = (char*)d_ws;
  size_t off = 0;
  auto alloc = [&](size_t bytes) { char* p = ws + off; off += rup(bytes); return p; };
  float*    cnt    = (float*)   alloc((size_t)N * 4);
  float*    sumat  = (float*)   alloc((size_t)N * 4);
  float*    loopat = (float*)   alloc((size_t)N * 4);
  __bf16*   xbf    = (__bf16*)  alloc((size_t)N * D * 2);
  __bf16*   hbf    = (__bf16*)  alloc((size_t)N * D * 2);
  __bf16*   WtA    = (__bf16*)  alloc((size_t)M2 * D * 2);
  __bf16*   WtB    = (__bf16*)  alloc((size_t)M2 * D * 2);
  float*    xl1    = (float*)   alloc((size_t)N * M1 * 4);
  float*    xr1    = (float*)   alloc((size_t)N * M1 * 4);
  float*    h1     = (float*)   alloc((size_t)N * M1 * 4);
  float*    xl2    = (float*)   alloc((size_t)N * M2 * 4);
  float*    xr2    = (float*)   alloc((size_t)N * M2 * 4);
  float*    logitsB= (float*)   alloc((size_t)Etot * H * 4);
  unsigned* mEnc   = (unsigned*)alloc((size_t)N * H * 4);
  float*    denom  = (float*)   alloc((size_t)N * H * 4);
  (void)ws_size; (void)n_in; (void)out_size;

  float* outF     = (float*)d_out;               // [N, C2]
  float* alphaOut = outF + (size_t)N * C2;       // [Etot, H]

  const int B = 256;
  dim3 blk(B);

  // ---- self loop attrs ----
  hipMemsetAsync(cnt,   0, (size_t)N * 4, stream);
  hipMemsetAsync(sumat, 0, (size_t)N * 4, stream);
  gat_degree_kernel<<<cdiv(E, B), blk, 0, stream>>>(ei, ea, cnt, sumat, E);
  gat_loopattr_kernel<<<cdiv(N, B), blk, 0, stream>>>(cnt, sumat, loopat, N);

  // ---- layer 1 GEMMs (bf16 WMMA, LDS-staged weights) ----
  gat_cvt_bf16_kernel<<<cdiv((long long)N * D, B), blk, 0, stream>>>(x, xbf, N * D);
  gat_transpose_cvt_kernel<<<cdiv((long long)D * M1, B), blk, 0, stream>>>(Wl1, WtA, D, M1);
  gat_transpose_cvt_kernel<<<cdiv((long long)D * M1, B), blk, 0, stream>>>(Wr1, WtB, D, M1);
  {
    int blocks = (N / GROWS) * (M1 / GCOLS);
    gat_gemm_wmma_bf16<<<blocks, blk, 0, stream>>>(xbf, WtA, bl1, xl1, D, M1);
    gat_gemm_wmma_bf16<<<blocks, blk, 0, stream>>>(xbf, WtB, br1, xr1, D, M1);
  }

  // ---- layer 1 attention + aggregation ----
  hipMemsetAsync(mEnc,  0, (size_t)N * H * 4, stream);
  hipMemsetAsync(denom, 0, (size_t)N * H * 4, stream);
  hipMemsetAsync(h1,    0, (size_t)N * M1 * 4, stream);
  gat_logits_kernel<<<cdiv((long long)Etot * 32, B), blk, 0, stream>>>(
      xl1, xr1, ei, ea, loopat, We1, att1, logitsB, mEnc, E, N, H, C1);
  gat_exp_kernel<<<cdiv((long long)Etot * H, B), blk, 0, stream>>>(
      logitsB, ei, mEnc, logitsB, denom, E, N, H);
  gat_alpha_kernel<<<cdiv((long long)Etot * H, B), blk, 0, stream>>>(
      logitsB, ei, denom, E, N, H);
  gat_aggregate_kernel<<<cdiv((long long)Etot * 32, B), blk, 0, stream>>>(
      xl1, logitsB, ei, h1, E, N, H, C1, /*concat=*/1);
  gat_bias_relu_kernel<<<cdiv((long long)N * M1, B), blk, 0, stream>>>(h1, b1, N * M1, M1);

  // ---- layer 2 GEMMs ----
  gat_cvt_bf16_kernel<<<cdiv((long long)N * D, B), blk, 0, stream>>>(h1, hbf, N * D);
  gat_transpose_cvt_kernel<<<cdiv((long long)D * M2, B), blk, 0, stream>>>(Wl2, WtA, D, M2);
  gat_transpose_cvt_kernel<<<cdiv((long long)D * M2, B), blk, 0, stream>>>(Wr2, WtB, D, M2);
  {
    int blocks = (N / GROWS) * (M2 / GCOLS);
    gat_gemm_wmma_bf16<<<blocks, blk, 0, stream>>>(hbf, WtA, bl2, xl2, D, M2);
    gat_gemm_wmma_bf16<<<blocks, blk, 0, stream>>>(hbf, WtB, br2, xr2, D, M2);
  }

  // ---- layer 2 attention + aggregation (alpha -> d_out tail) ----
  hipMemsetAsync(mEnc,  0, (size_t)N * H * 4, stream);
  hipMemsetAsync(denom, 0, (size_t)N * H * 4, stream);
  gat_init_out_kernel<<<cdiv((long long)N * C2, B), blk, 0, stream>>>(outF, b2, N * C2, C2);
  gat_logits_kernel<<<cdiv((long long)Etot * 32, B), blk, 0, stream>>>(
      xl2, xr2, ei, ea, loopat, We2, att2, logitsB, mEnc, E, N, H, C2);
  gat_exp_kernel<<<cdiv((long long)Etot * H, B), blk, 0, stream>>>(
      logitsB, ei, mEnc, alphaOut, denom, E, N, H);
  gat_alpha_kernel<<<cdiv((long long)Etot * H, B), blk, 0, stream>>>(
      alphaOut, ei, denom, E, N, H);
  gat_aggregate_kernel<<<cdiv((long long)Etot * 32, B), blk, 0, stream>>>(
      xl2, alphaOut, ei, outF, E, N, H, C2, /*concat=*/0);
}
